// E3nnEdgeMessageBlock_3358664425486
// MI455X (gfx1250) — matrix operand order repaired and checked by reading, not compiled, not run on hardware
//
#include <hip/hip_runtime.h>
#include <hip/hip_bf16.h>

// ---------------------------------------------------------------------------
// E3nn edge message block, fused for gfx1250 (wave32 + v_wmma_f32_16x16x32_bf16)
//
//  Kernel 0 (pack_b):  weights -> bf16 wave32 WMMA B-fragment layout
//                      (per (ktile,ntile,lane): 16 contiguous bf16 = 32B),
//                      1/sqrt(K)-style scales folded in.
//  Kernel 1 (node_up): s = nf[:, :128] @ W_up_s, v_m = nf_m @ W_up_v  (WMMA),
//                      bf16 nodebuf[node][512] = [s | v0 | v1 | v2].
//  Kernel 2 (edge):    per wave: 16 edges. gather node rows -> LDS, MLP
//                      (layer1 scalar, layers2/3 + W4 via WMMA), tensor product
//                      elementwise -> LDS A-tiles, then 5 accumulating WMMA
//                      GEMMs (s_out, p = (w_c*s1)@Wv_top shared across m,
//                      q_m = ((w_d*y0)*v1_m)@Wv_bot), f32 output, v stores as
//                      contiguous 12B triplets.
//  LDS: exactly 40KB/wave, 160KB/block -> 2 blocks per WGP.
// ---------------------------------------------------------------------------

typedef __bf16 bf16_t;
typedef __attribute__((ext_vector_type(16))) __bf16 bf16x16;
typedef __attribute__((ext_vector_type(8)))  __bf16 bf16x8;
typedef __attribute__((ext_vector_type(8)))  float  f32x8;

#define SILU_CST   1.6792f        // 1/sqrt(E[silu(z)^2]), z~N(0,1)
#define INV_SQRT3  0.57735027f
#define INV_SQRT8  0.35355339f

struct F3 { float x, y, z; };     // 12B packed, 4B aligned -> contiguous stores

__device__ __forceinline__ f32x8 wmma_bf(bf16x16 a, bf16x16 b, f32x8 c) {
  return __builtin_amdgcn_wmma_f32_16x16x32_bf16(false, a, false, b, (short)0, c,
                                                 false, false);
}

// A fragment (16x32 bf16) from row-major LDS tile; rs = row stride in elems.
// lane<16: row=lane, K = {k0..k0+7, k0+16..k0+23}; lane>=16: K shifted by 8.
__device__ __forceinline__ bf16x16 lds_a_frag(const bf16_t* row0, int rs, int lane) {
  const bf16_t* p = row0 + (lane & 15) * rs + ((lane >> 4) << 3);
  bf16x8 lo = *(const bf16x8*)(p);
  bf16x8 hi = *(const bf16x8*)(p + 16);
  bf16x16 a;
#pragma unroll
  for (int i = 0; i < 8; ++i) { a[i] = lo[i]; a[i + 8] = hi[i]; }
  return a;
}

// B fragment from pre-packed global weights: 32B contiguous per lane.
__device__ __forceinline__ bf16x16 g_b_frag(const bf16_t* pack, int tile, int lane) {
  return *(const bf16x16*)(pack + (((tile << 5) + lane) << 4));
}

__device__ __forceinline__ float silu_n(float x) {
  return x / (1.0f + __expf(-x)) * SILU_CST;
}

// ---------------------------------------------------------------------------
// Kernel 0: pack W (KxN f32, row-major) -> bf16 B-fragment layout, *scale.
// ---------------------------------------------------------------------------
__global__ void pack_b_kernel(const float* __restrict__ W, bf16_t* __restrict__ dst,
                              int K, int N, float scale) {
  int o = blockIdx.x * blockDim.x + threadIdx.x;
  if (o >= K * N) return;
  int i    = o & 15;
  int lane = (o >> 4) & 31;
  int tile = o >> 9;
  int NT   = N >> 4;
  int nt   = tile % NT;
  int kt   = tile / NT;
  int k    = kt * 32 + ((lane & 16) ? 8 : 0) + (i & 7) + ((i & 8) ? 16 : 0);
  int n    = nt * 16 + (lane & 15);
  dst[o] = (bf16_t)(W[k * N + n] * scale);
}

// ---------------------------------------------------------------------------
// Kernel 1: node up-projection. 4 waves x 16 nodes per block.
// nodebuf[node][0:128]=s, [128+128m : 256+128m] = v_m   (bf16)
// N_NODES % 16 == 0 -> the store guard is wave-uniform (no EXEC churn).
// ---------------------------------------------------------------------------
__global__ void __launch_bounds__(128)
node_up_kernel(const float* __restrict__ nf, const bf16_t* __restrict__ packS,
               const bf16_t* __restrict__ packV, bf16_t* __restrict__ nodebuf,
               int n_nodes) {
  __shared__ bf16_t buf[4][16 * 128];
  int lane = threadIdx.x & 31;
  int w    = threadIdx.x >> 5;
  int nb   = blockIdx.x * 64 + w * 16;
  bool live = nb < n_nodes;                 // uniform per wave (16 | n_nodes)
  bf16_t* B = buf[w];

  for (int p = 0; p < 4; ++p) {
    for (int t = 0; t < 64; ++t) {
      int idx = t * 32 + lane;
      int row = idx >> 7, col = idx & 127;
      int node = nb + row; if (node >= n_nodes) node = n_nodes - 1;
      float x = (p == 0) ? nf[(size_t)node * 512 + col]
                         : nf[(size_t)node * 512 + 128 + col * 3 + (p - 1)];
      B[row * 128 + col] = (bf16_t)x;
    }
    bf16x16 af[4];
#pragma unroll
    for (int k = 0; k < 4; ++k) af[k] = lds_a_frag(B + k * 32, 128, lane);
    const bf16_t* pk = (p == 0) ? packS : packV;
#pragma unroll
    for (int nt = 0; nt < 8; ++nt) {
      f32x8 acc = {};
#pragma unroll
      for (int k = 0; k < 4; ++k) acc = wmma_bf(af[k], g_b_frag(pk, k * 8 + nt, lane), acc);
      if (live) {                            // uniform branch
        int colb = (p == 0) ? nt * 16 : 128 + (p - 1) * 128 + nt * 16;
        int n = colb + (lane & 15);
#pragma unroll
        for (int r = 0; r < 8; ++r) {
          int node = nb + r + ((lane >> 4) << 3);
          nodebuf[(size_t)node * 512 + n] = (bf16_t)acc[r];
        }
      }
    }
  }
}

// ---------------------------------------------------------------------------
// Kernel 2: fused edge pipeline. 4 waves/block, 16 edges/wave.
// Per-wave LDS (bf16 elems): sv[16][512] | ab[16][768]  == 20480 elems = 40KB.
// hb (MLP acts, 1024 elems) and sidx (16 ints) alias the front of ab: their
// lifetimes end before ab is first written; same-wave LDS ops are in-order.
// ---------------------------------------------------------------------------
#define WAVE_ELEMS 20480

__global__ void __launch_bounds__(128, 1)
edge_kernel(const long long* __restrict__ eidx, const float* __restrict__ eattr,
            const float* __restrict__ efeat,   const float* __restrict__ W1,
            const bf16_t* __restrict__ nb,
            const bf16_t* __restrict__ pW2,  const bf16_t* __restrict__ pW3,
            const bf16_t* __restrict__ pW4,  const bf16_t* __restrict__ pWS,
            const bf16_t* __restrict__ pWVt, const bf16_t* __restrict__ pWVb,
            float* __restrict__ out) {
  extern __shared__ bf16_t sm[];
  int lane  = threadIdx.x & 31;
  int w     = threadIdx.x >> 5;
  int rbase = (lane >> 4) << 3;             // rows 0..7 or 8..15
  bf16_t* base = sm + w * WAVE_ELEMS;
  bf16_t* sv   = base;                      // 16 x 512
  bf16_t* ab   = base + 8192;               // 16 x 768 : [o0|o1|cs|dv0|dv1|dv2]
  bf16_t* hb   = ab;                        // alias: 16 x 64 (dead before ab use)
  int*    sidx = (int*)ab;                  // alias: 16 ints (dead before hb use)
  int e0 = (blockIdx.x * 4 + w) * 16;

  // --- sender indices into LDS; edge attrs into registers (own 8 rows) ---
  if (lane < 16) sidx[lane] = (int)eidx[e0 + lane];
  float y0r[8], y1r[8][3];
#pragma unroll
  for (int rr = 0; rr < 8; ++rr) {
    const float4 a = *(const float4*)&eattr[(size_t)(e0 + rbase + rr) * 4];
    y0r[rr] = a.x; y1r[rr][0] = a.y; y1r[rr][1] = a.z; y1r[rr][2] = a.w;
  }

  // --- gather 16 node rows (1KB bf16 each): 2 lanes per row, 512B each ---
  {
    int row  = lane >> 1;
    int node = sidx[row];
    const char* src = (const char*)nb + (size_t)node * 1024 + (lane & 1) * 512;
    char*       dst = (char*)sv + row * 1024 + (lane & 1) * 512;
#pragma unroll
    for (int t = 0; t < 32; ++t)
      *(uint4*)(dst + t * 16) = *(const uint4*)(src + t * 16);
  }

  // --- MLP layer 1 (K=8, scalar): h1 = silu(ef @ W1/sqrt(8))*CST -> hb ---
  {
    int j0 = (lane & 15) * 4;
    float4 wcol[8];
#pragma unroll
    for (int k = 0; k < 8; ++k) wcol[k] = *(const float4*)&W1[k * 64 + j0];
#pragma unroll
    for (int rr = 0; rr < 8; ++rr) {
      int m = rbase + rr;
      float4 xa = *(const float4*)&efeat[(size_t)(e0 + m) * 8];
      float4 xb = *(const float4*)&efeat[(size_t)(e0 + m) * 8 + 4];
      float x[8] = {xa.x, xa.y, xa.z, xa.w, xb.x, xb.y, xb.z, xb.w};
      float acc0 = 0.f, acc1 = 0.f, acc2 = 0.f, acc3 = 0.f;
#pragma unroll
      for (int k = 0; k < 8; ++k) {
        acc0 += x[k] * wcol[k].x; acc1 += x[k] * wcol[k].y;
        acc2 += x[k] * wcol[k].z; acc3 += x[k] * wcol[k].w;
      }
      hb[m * 64 + j0 + 0] = (bf16_t)silu_n(acc0 * INV_SQRT8);
      hb[m * 64 + j0 + 1] = (bf16_t)silu_n(acc1 * INV_SQRT8);
      hb[m * 64 + j0 + 2] = (bf16_t)silu_n(acc2 * INV_SQRT8);
      hb[m * 64 + j0 + 3] = (bf16_t)silu_n(acc3 * INV_SQRT8);
    }
  }

  // --- MLP layers 2 & 3 via WMMA (1/sqrt(64) folded into packs) ---
#pragma unroll
  for (int layer = 0; layer < 2; ++layer) {
    const bf16_t* pw = layer ? pW3 : pW2;
    bf16x16 af0 = lds_a_frag(hb, 64, lane);
    bf16x16 af1 = lds_a_frag(hb + 32, 64, lane);
    f32x8 acc[4];
#pragma unroll
    for (int nt = 0; nt < 4; ++nt) {
      f32x8 z = {};
      z = wmma_bf(af0, g_b_frag(pw, nt, lane), z);
      z = wmma_bf(af1, g_b_frag(pw, 4 + nt, lane), z);
      acc[nt] = z;
    }
#pragma unroll
    for (int nt = 0; nt < 4; ++nt) {
      int n = nt * 16 + (lane & 15);
#pragma unroll
      for (int r = 0; r < 8; ++r)
        hb[(rbase + r) * 64 + n] = (bf16_t)silu_n(acc[nt][r]);
    }
  }

  // --- tpw = h3 @ W4' (16x512), fused tensor product -> A tiles in ab ---
  bf16x16 h3f0 = lds_a_frag(hb, 64, lane);
  bf16x16 h3f1 = lds_a_frag(hb + 32, 64, lane);
#pragma unroll
  for (int t = 0; t < 32; ++t) {
    f32x8 z = {};
    z = wmma_bf(h3f0, g_b_frag(pW4, t, lane), z);
    z = wmma_bf(h3f1, g_b_frag(pW4, 32 + t, lane), z);
    int g = t >> 3;                   // 0:w_a 1:w_b 2:w_c 3:w_d
    int u = ((t & 7) << 4) + (lane & 15);
#pragma unroll
    for (int r = 0; r < 8; ++r) {
      int m = rbase + r;
      float wv = z[r];
      if (g == 0) {
        float s1 = (float)sv[m * 512 + u];
        ab[m * 768 + u] = (bf16_t)(wv * s1 * y0r[r]);
      } else if (g == 1) {
        float d = 0.f;
#pragma unroll
        for (int mm = 0; mm < 3; ++mm)
          d += (float)sv[m * 512 + 128 + mm * 128 + u] * y1r[r][mm];
        ab[m * 768 + 128 + u] = (bf16_t)(wv * d * INV_SQRT3);
      } else if (g == 2) {
        float s1 = (float)sv[m * 512 + u];
        ab[m * 768 + 256 + u] = (bf16_t)(wv * s1);
      } else {
        float wd0 = wv * y0r[r];
#pragma unroll
        for (int mm = 0; mm < 3; ++mm) {
          float v1 = (float)sv[m * 512 + 128 + mm * 128 + u];
          ab[m * 768 + 384 + mm * 128 + u] = (bf16_t)(wd0 * v1);
        }
      }
    }
  }

  // --- s_out = [o0|o1] @ W_out_s' ---
  {
    f32x8 acc[8];
#pragma unroll
    for (int n = 0; n < 8; ++n) { f32x8 zz = {}; acc[n] = zz; }
#pragma unroll
    for (int k = 0; k < 8; ++k) {
      bf16x16 a = lds_a_frag(ab + k * 32, 768, lane);
#pragma unroll
      for (int n = 0; n < 8; ++n)
        acc[n] = wmma_bf(a, g_b_frag(pWS, k * 8 + n, lane), acc[n]);
    }
#pragma unroll
    for (int n = 0; n < 8; ++n) {
      int col = n * 16 + (lane & 15);
#pragma unroll
      for (int r = 0; r < 8; ++r)
        out[(size_t)(e0 + rbase + r) * 512 + col] = acc[n][r];
    }
  }

  // --- v_out_m = y1_m * ((w_c*s1)@Wv_top) + ((w_d*y0)*v1_m)@Wv_bot ---
  {
    f32x8 accP[8];
#pragma unroll
    for (int n = 0; n < 8; ++n) { f32x8 zz = {}; accP[n] = zz; }
#pragma unroll
    for (int k = 0; k < 4; ++k) {
      bf16x16 a = lds_a_frag(ab + 256 + k * 32, 768, lane);
#pragma unroll
      for (int n = 0; n < 8; ++n)
        accP[n] = wmma_bf(a, g_b_frag(pWVt, k * 8 + n, lane), accP[n]);
    }
    // all three q_m tiles live simultaneously -> 12B-contiguous stores
    f32x8 accQ[3][8];
#pragma unroll
    for (int mm = 0; mm < 3; ++mm)
#pragma unroll
      for (int n = 0; n < 8; ++n) { f32x8 zz = {}; accQ[mm][n] = zz; }
#pragma unroll
    for (int mm = 0; mm < 3; ++mm)
#pragma unroll
      for (int k = 0; k < 4; ++k) {
        bf16x16 a = lds_a_frag(ab + 384 + mm * 128 + k * 32, 768, lane);
#pragma unroll
        for (int n = 0; n < 8; ++n)
          accQ[mm][n] = wmma_bf(a, g_b_frag(pWVb, k * 8 + n, lane), accQ[mm][n]);
      }
#pragma unroll
    for (int n = 0; n < 8; ++n) {
      int u = n * 16 + (lane & 15);
#pragma unroll
      for (int r = 0; r < 8; ++r) {
        float px = accP[n][r];
        F3 v;
        v.x = y1r[r][0] * px + accQ[0][n][r];
        v.y = y1r[r][1] * px + accQ[1][n][r];
        v.z = y1r[r][2] * px + accQ[2][n][r];
        *(F3*)(out + (size_t)(e0 + rbase + r) * 512 + 128 + (size_t)u * 3) = v;
      }
    }
  }
}

// ---------------------------------------------------------------------------
extern "C" void kernel_launch(void* const* d_in, const int* in_sizes, int n_in,
                              void* d_out, int out_size, void* d_ws, size_t ws_size,
                              hipStream_t stream) {
  const float*     node_feats = (const float*)d_in[0];
  const long long* edge_index = (const long long*)d_in[1];
  const float*     edge_attrs = (const float*)d_in[2];
  const float*     edge_feats = (const float*)d_in[3];
  const float*     W_up_s     = (const float*)d_in[4];
  const float*     W_up_v     = (const float*)d_in[5];
  const float*     W1         = (const float*)d_in[6];
  const float*     W2         = (const float*)d_in[7];
  const float*     W3         = (const float*)d_in[8];
  const float*     W4         = (const float*)d_in[9];
  const float*     W_out_s    = (const float*)d_in[10];
  const float*     W_out_v    = (const float*)d_in[11];
  float* out = (float*)d_out;

  // workspace layout (bytes); total ~10.5 MB
  char* ws = (char*)d_ws;
  bf16_t* nodebuf = (bf16_t*)ws;                       // 10000*512*2
  size_t o = 10240000;
  bf16_t* pW2  = (bf16_t*)(ws + o); o += 64 * 64 * 2;
  bf16_t* pW3  = (bf16_t*)(ws + o); o += 64 * 64 * 2;
  bf16_t* pW4  = (bf16_t*)(ws + o); o += 64 * 512 * 2;
  bf16_t* pWS  = (bf16_t*)(ws + o); o += 256 * 128 * 2;
  bf16_t* pWVt = (bf16_t*)(ws + o); o += 128 * 128 * 2;
  bf16_t* pWVb = (bf16_t*)(ws + o); o += 128 * 128 * 2;
  bf16_t* pUpS = (bf16_t*)(ws + o); o += 128 * 128 * 2;
  bf16_t* pUpV = (bf16_t*)(ws + o); o += 128 * 128 * 2;

  const float inv_sqrt_mul  = 0.088388348f;   // 1/sqrt(128)
  const float inv_sqrt_2mul = 0.0625f;        // 1/sqrt(256)
  const float inv64         = 0.125f;         // 1/sqrt(64)

  pack_b_kernel<<<(64 * 64 + 255) / 256, 256, 0, stream>>>(W2, pW2, 64, 64, inv64);
  pack_b_kernel<<<(64 * 64 + 255) / 256, 256, 0, stream>>>(W3, pW3, 64, 64, inv64);
  pack_b_kernel<<<(64 * 512 + 255) / 256, 256, 0, stream>>>(W4, pW4, 64, 512, inv64);
  pack_b_kernel<<<(256 * 128 + 255) / 256, 256, 0, stream>>>(W_out_s, pWS, 256, 128, inv_sqrt_2mul);
  pack_b_kernel<<<(128 * 128 + 255) / 256, 256, 0, stream>>>(W_out_v, pWVt, 128, 128, inv_sqrt_2mul);
  pack_b_kernel<<<(128 * 128 + 255) / 256, 256, 0, stream>>>(W_out_v + 128 * 128, pWVb, 128, 128, inv_sqrt_2mul);
  pack_b_kernel<<<(128 * 128 + 255) / 256, 256, 0, stream>>>(W_up_s, pUpS, 128, 128, inv_sqrt_mul);
  pack_b_kernel<<<(128 * 128 + 255) / 256, 256, 0, stream>>>(W_up_v, pUpV, 128, 128, inv_sqrt_mul);

  int n_nodes = 10000;
  node_up_kernel<<<(n_nodes + 63) / 64, 128, 0, stream>>>(node_feats, pUpS, pUpV,
                                                          nodebuf, n_nodes);

  // 64 edges/block, 200000/64 = 3125 blocks; 160KB LDS -> 2 blocks/WGP
  size_t smem = (size_t)4 * WAVE_ELEMS * sizeof(bf16_t);  // 163,840 B
  edge_kernel<<<3125, 128, smem, stream>>>(edge_index, edge_attrs, edge_feats, W1,
                                           nodebuf, pW2, pW3, pW4, pWS, pWVt, pWVb,
                                           out);
}
